// hashNerf_21646635172584
// MI455X (gfx1250) — matrix operand order repaired and checked by reading, not compiled, not run on hardware
//
#include <hip/hip_runtime.h>

typedef __attribute__((ext_vector_type(16))) _Float16 v16h;
typedef __attribute__((ext_vector_type(8)))  float    v8f;

#define NPTS   1048576
#define TSIZE  262144
#define TMASK  0x3FFFFu
#define PRIME  2654435761u
#define WAVES  8

union V16HQ { uint4 q[2]; v16h v; };

// A-matrix (16x32 f16) fragment: row = point row base pointer (+koff for K 32..63).
// lanes 0-15: elems = halfs [kb..kb+8) and [kb+16..kb+24) with kb=0; lanes 16-31: kb=8.
static __device__ __forceinline__ v16h ldsA(const _Float16* row, int kb) {
  V16HQ u;
  u.q[0] = *(const uint4*)(row + kb);
  u.q[1] = *(const uint4*)(row + kb + 16);
  return u.v;
}
// B-matrix (32x16 f16) fragment from transposed weights Wt[n][k]:
// lane's 16 halfs are K=[kb..kb+8) and [kb+16..kb+24) (kb = 0 for lanes 0-15, 8 for lanes 16-31).
static __device__ __forceinline__ v16h ldsB(const _Float16* p) {
  V16HQ u;
  u.q[0] = *(const uint4*)(p);
  u.q[1] = *(const uint4*)(p + 16);
  return u.v;
}
static __device__ __forceinline__ v8f splat8(float x) {
  v8f c;
#pragma unroll
  for (int j = 0; j < 8; ++j) c[j] = x;
  return c;
}
static __device__ __forceinline__ v8f leaky8(v8f c) {
#pragma unroll
  for (int j = 0; j < 8; ++j) c[j] = fmaxf(c[j], 0.01f * c[j]);
  return c;
}
#define WMMA(A, B, C) __builtin_amdgcn_wmma_f32_16x16x32_f16(false, (A), false, (B), (short)0, (C), false, false)

__global__ __launch_bounds__(256)
void hashnerf_fused(const float* __restrict__ X,
                    const float* __restrict__ HT,
                    const float* __restrict__ W1, const float* __restrict__ b1,
                    const float* __restrict__ W2, const float* __restrict__ b2,
                    const float* __restrict__ W3, const float* __restrict__ b3,
                    const float* __restrict__ W4, const float* __restrict__ b4,
                    float* __restrict__ out)
{
  __shared__ __align__(16) _Float16 sW1[64 * 32];   // W1^T [n][k]
  __shared__ __align__(16) _Float16 sW2[64 * 64];   // W2^T [n][k]
  __shared__ __align__(16) _Float16 sW3[64 * 64];   // W3^T [n][k]
  __shared__ __align__(16) _Float16 sW4[16 * 64];   // W4^T padded to 16 out-ch
  __shared__ float sB1[64], sB2[64], sB3[64], sB4[16];
  __shared__ __align__(16) _Float16 sFeat[WAVES][16 * 32];
  __shared__ __align__(16) _Float16 sH[WAVES][16 * 64];

  const int tid = threadIdx.x;

  // ---- cooperative weight preload: fp32 global -> fp16 transposed LDS ----
  for (int i = tid; i < 64 * 32; i += 256) {
    int n = i >> 5, k = i & 31;
    sW1[i] = (_Float16)W1[k * 64 + n];
  }
  for (int i = tid; i < 64 * 64; i += 256) {
    int n = i >> 6, k = i & 63;
    sW2[i] = (_Float16)W2[k * 64 + n];
    sW3[i] = (_Float16)W3[k * 64 + n];
  }
  for (int i = tid; i < 16 * 64; i += 256) {
    int n = i >> 6, k = i & 63;
    sW4[i] = (n < 3) ? (_Float16)W4[k * 3 + n] : (_Float16)0.0f;
  }
  if (tid < 64) { sB1[tid] = b1[tid]; sB2[tid] = b2[tid]; sB3[tid] = b3[tid]; }
  if (tid < 16) sB4[tid] = (tid < 3) ? b4[tid] : 0.0f;
  __syncthreads();

  const int wave = tid >> 5;
  const int lane = tid & 31;
  const int prow = lane & 15;       // point row within tile / N column for C/D
  const int hi   = lane >> 4;       // half-wave select
  const int moff = hi * 8;          // C/D row offset for lanes 16-31
  const int kbA  = hi * 8;          // A-fragment K base
  const int kbB  = hi * 8;          // B-fragment K base
  const int ncol = prow;

  const int tile = blockIdx.x * WAVES + wave;
  const int p0   = tile * 16;
  const int p    = p0 + prow;

  // ================= hash-grid encoding (16 pts x 16 levels) =================
  const float x = X[2 * p], y = X[2 * p + 1];
  _Float16* feat = sFeat[wave];
  const int lbase = hi * 8;
  const float NVlo[8] = {16.f, 21.f, 27.f, 36.f, 48.f, 64.f, 84.f, 111.f};
  const float NVhi[8] = {147.f, 194.f, 256.f, 337.f, 445.f, 588.f, 776.f, 1024.f};
#pragma unroll
  for (int i = 0; i < 8; ++i) {
    const float Nl = hi ? NVhi[i] : NVlo[i];
    const int l = lbase + i;
    float xs = x * Nl, ys = y * Nl;
    float xf = floorf(xs), yf = floorf(ys);
    unsigned xi = (unsigned)xf;
    unsigned hy = (unsigned)yf * PRIME;
    unsigned h01 = hy & TMASK;            // corner (0, yf)
    unsigned h10 = xi & TMASK;            // corner (xf, 0)
    unsigned h11 = (xi ^ hy) & TMASK;     // corner (xf, yf)
    const float2* tb = (const float2*)HT + (size_t)l * TSIZE;
    float2 v00 = tb[0];                   // corner (0,0) hashes to slot 0
    float2 v01 = tb[h01];
    float2 v10 = tb[h10];
    float2 v11 = tb[h11];
    float fx = xs - xf, fy = ys - yf;
    float cx = 1.f - fx, cy = 1.f - fy;
    float w00 = cx * cy, w01 = cx * fy, w10 = fx * cy, w11 = fx * fy;
    float f0 = w00 * v00.x + w01 * v01.x + w10 * v10.x + w11 * v11.x;
    float f1 = w00 * v00.y + w01 * v01.y + w10 * v10.y + w11 * v11.y;
    union { _Float16 h[2]; unsigned u; } pk;
    pk.h[0] = (_Float16)f0; pk.h[1] = (_Float16)f1;
    *(unsigned*)(feat + prow * 32 + l * 2) = pk.u;   // packed b32 LDS store
  }

  _Float16* hb = sH[wave];

  // ================= layer 1: feat(16x32) @ W1 + b1, leaky =================
  {
    v16h a0 = ldsA(feat + prow * 32, kbA);
    v8f acc[4];
#pragma unroll
    for (int t = 0; t < 4; ++t) {
      const int n = ncol + 16 * t;
      v8f c = splat8(sB1[n]);
      v16h b0 = ldsB(sW1 + n * 32 + kbB);
      c = WMMA(a0, b0, c);
      acc[t] = leaky8(c);
    }
#pragma unroll
    for (int t = 0; t < 4; ++t)
#pragma unroll
      for (int j = 0; j < 8; ++j)
        hb[(j + moff) * 64 + ncol + 16 * t] = (_Float16)acc[t][j];
  }
  // ================= layer 2: h(16x64) @ W2 + b2, leaky =================
  {
    v16h a0 = ldsA(hb + prow * 64, kbA);
    v16h a1 = ldsA(hb + prow * 64 + 32, kbA);
    v8f acc[4];
#pragma unroll
    for (int t = 0; t < 4; ++t) {
      const int n = ncol + 16 * t;
      v8f c = splat8(sB2[n]);
      v16h b0 = ldsB(sW2 + n * 64 + kbB);
      v16h b1 = ldsB(sW2 + n * 64 + 32 + kbB);
      c = WMMA(a0, b0, c);
      c = WMMA(a1, b1, c);
      acc[t] = leaky8(c);
    }
#pragma unroll
    for (int t = 0; t < 4; ++t)
#pragma unroll
      for (int j = 0; j < 8; ++j)
        hb[(j + moff) * 64 + ncol + 16 * t] = (_Float16)acc[t][j];
  }
  // ================= layer 3: h(16x64) @ W3 + b3, leaky =================
  {
    v16h a0 = ldsA(hb + prow * 64, kbA);
    v16h a1 = ldsA(hb + prow * 64 + 32, kbA);
    v8f acc[4];
#pragma unroll
    for (int t = 0; t < 4; ++t) {
      const int n = ncol + 16 * t;
      v8f c = splat8(sB3[n]);
      v16h b0 = ldsB(sW3 + n * 64 + kbB);
      v16h b1 = ldsB(sW3 + n * 64 + 32 + kbB);
      c = WMMA(a0, b0, c);
      c = WMMA(a1, b1, c);
      acc[t] = leaky8(c);
    }
#pragma unroll
    for (int t = 0; t < 4; ++t)
#pragma unroll
      for (int j = 0; j < 8; ++j)
        hb[(j + moff) * 64 + ncol + 16 * t] = (_Float16)acc[t][j];
  }
  // ================= layer 4: h(16x64) @ W4 + b4, relu, store =================
  {
    v16h a0 = ldsA(hb + prow * 64, kbA);
    v16h a1 = ldsA(hb + prow * 64 + 32, kbA);
    v8f c = splat8(sB4[ncol]);
    v16h b0 = ldsB(sW4 + ncol * 64 + kbB);
    v16h b1 = ldsB(sW4 + ncol * 64 + 32 + kbB);
    c = WMMA(a0, b0, c);
    c = WMMA(a1, b1, c);
#pragma unroll
    for (int j = 0; j < 8; ++j) c[j] = fmaxf(c[j], 0.0f);
    if (ncol < 3) {
#pragma unroll
      for (int j = 0; j < 8; ++j)
        out[(size_t)(p0 + j + moff) * 3 + ncol] = c[j];
    }
  }
}

extern "C" void kernel_launch(void* const* d_in, const int* in_sizes, int n_in,
                              void* d_out, int out_size, void* d_ws, size_t ws_size,
                              hipStream_t stream) {
  const float* X  = (const float*)d_in[0];
  const float* HT = (const float*)d_in[1];
  const float* W1 = (const float*)d_in[2];
  const float* b1 = (const float*)d_in[3];
  const float* W2 = (const float*)d_in[4];
  const float* b2 = (const float*)d_in[5];
  const float* W3 = (const float*)d_in[6];
  const float* b3 = (const float*)d_in[7];
  const float* W4 = (const float*)d_in[8];
  const float* b4 = (const float*)d_in[9];
  float* out = (float*)d_out;

  dim3 grid(NPTS / (16 * WAVES));   // 8192 blocks: 8 waves/block, 16 pts/wave
  dim3 block(256);
  hipLaunchKernelGGL(hashnerf_fused, grid, block, 0, stream,
                     X, HT, W1, b1, W2, b2, W3, b3, W4, b4, out);
}
